// StarTransformerDecoderLayer_1460288881173
// MI455X (gfx1250) — matrix operand order repaired and checked by reading, not compiled, not run on hardware
//
#include <hip/hip_runtime.h>
#include <stdint.h>
#include <stddef.h>

// ---------------- problem constants (from reference setup) ----------------
#define BB   8
#define LLN  1024
#define DDIM 512
#define NHH  8
#define DH   64
#define DFFN 2048
#define BLR  (BB*LLN)          // 8192 rows

typedef __attribute__((ext_vector_type(16))) __bf16 v16bf;
typedef __attribute__((ext_vector_type(8)))  float  v8f;

__device__ __forceinline__ uint16_t f2bf(float x) {
  uint32_t u = __float_as_uint(x);
  uint32_t r = (u + 0x7FFFu + ((u >> 16) & 1u)) >> 16;  // RNE
  return (uint16_t)r;
}

// =============================================================================
// Tiled GEMM: C[M,N] = act(A[M,K] @ W[K,N] + bias)
// fp32 in/out, bf16 WMMA compute with f32 accumulation.
// Block = 256 threads (8 wave32), 128x128 tile, K-step 32.
// Wave w: M-strip 32*(w&3), N-strip 64*(w>>2)  -> 2x4 of 16x16 WMMA tiles.
// LDS: A tile row-major [128][32], W tile as [n][k] so both A and B fragments
// load contiguous bf16 K-pairs with single b32 ds reads (contiguous k = 16*g + s in A and B).
// =============================================================================
template <bool RELU>
__global__ __launch_bounds__(256) void gemm_wmma(
    const float* __restrict__ A, const float* __restrict__ W,
    const float* __restrict__ bias, float* __restrict__ C,
    int M, int N, int K)
{
  __shared__ uint16_t As[128 * 34];
  __shared__ uint16_t Ws[128 * 34];
  const int tid  = threadIdx.x;
  const int lane = tid & 31;
  const int wave = tid >> 5;
  const int wr   = wave & 3;     // M strip
  const int wc   = wave >> 2;    // N strip
  const int bm   = blockIdx.y * 128;
  const int bn   = blockIdx.x * 128;
  const int g    = lane >> 4;    // lane half
  const int ln   = lane & 15;

  v8f acc[2][4];
#pragma unroll
  for (int i = 0; i < 2; ++i)
#pragma unroll
    for (int j = 0; j < 4; ++j)
#pragma unroll
      for (int r = 0; r < 8; ++r) acc[i][j][r] = 0.0f;

  for (int k0 = 0; k0 < K; k0 += 32) {
    // A tile 128x32 -> As[r][c], 16 elems/thread
#pragma unroll
    for (int i = 0; i < 16; ++i) {
      int idx = tid + 256 * i;
      int r = idx >> 5, c = idx & 31;
      int gm = bm + r;
      float v = (gm < M) ? A[(size_t)gm * K + (k0 + c)] : 0.0f;
      As[r * 34 + c] = f2bf(v);
    }
    // W tile 32x128 -> Ws[n][k] (transposed store, coalesced global read)
#pragma unroll
    for (int i = 0; i < 16; ++i) {
      int idx = tid + 256 * i;
      int kk = idx >> 7, n = idx & 127;
      int gn = bn + n;
      float v = (gn < N) ? W[(size_t)(k0 + kk) * N + gn] : 0.0f;
      Ws[n * 34 + kk] = f2bf(v);
    }
    __syncthreads();

    v16bf afr[2], bfr[4];
#pragma unroll
    for (int mt = 0; mt < 2; ++mt) {
      int m = 32 * wr + 16 * mt + ln;
      uint32_t* u = reinterpret_cast<uint32_t*>(&afr[mt]);
#pragma unroll
      for (int v = 0; v < 8; ++v) {
        int kk = 2 * v + 16 * g;  // contiguous k = 16*g + s
        u[v] = *reinterpret_cast<const uint32_t*>(&As[m * 34 + kk]);
      }
    }
#pragma unroll
    for (int nt = 0; nt < 4; ++nt) {
      int n = 64 * wc + 16 * nt + ln;
      uint32_t* u = reinterpret_cast<uint32_t*>(&bfr[nt]);
#pragma unroll
      for (int v = 0; v < 8; ++v) {
        int kk = 2 * v + 16 * g;                               // same k order as A
        u[v] = *reinterpret_cast<const uint32_t*>(&Ws[n * 34 + kk]);
      }
    }
#pragma unroll
    for (int mt = 0; mt < 2; ++mt)
#pragma unroll
      for (int nt = 0; nt < 4; ++nt)
        acc[mt][nt] = __builtin_amdgcn_wmma_f32_16x16x32_bf16(
            false, afr[mt], false, bfr[nt], (short)0, acc[mt][nt], false, false);
    __syncthreads();
  }

  // epilogue: C/D layout -> (m = r + 8*g, n = ln) within each 16x16 tile
#pragma unroll
  for (int mt = 0; mt < 2; ++mt)
#pragma unroll
    for (int nt = 0; nt < 4; ++nt)
#pragma unroll
      for (int r = 0; r < 8; ++r) {
        int gm = bm + 32 * wr + 16 * mt + r + 8 * g;
        int gn = bn + 64 * wc + 16 * nt + ln;
        if (gm < M && gn < N) {
          float v = acc[mt][nt][r];
          if (bias) v += bias[gn];
          if (RELU) v = fmaxf(v, 0.0f);
          C[(size_t)gm * N + gn] = v;
        }
      }
}

// =============================================================================
// Causal flash attention, one (b,head,q-tile64) per 128-thread block (4 waves).
// Q/K in LDS row-major (A / [n][k] B layouts), V transposed [dh][key].
// Online softmax per 16-row wave strip, P round-trips LDS as bf16.
// =============================================================================
__global__ __launch_bounds__(128) void flash_attn_causal(
    const float* __restrict__ Q, const float* __restrict__ K,
    const float* __restrict__ V, float* __restrict__ O)
{
  __shared__ uint16_t Qs[64 * 66];
  __shared__ uint16_t Ks[64 * 66];
  __shared__ uint16_t Vs[64 * 66];  // [dh][key]
  __shared__ uint16_t Ps[64 * 66];
  const int tid = threadIdx.x, lane = tid & 31, wave = tid >> 5;
  const int g = lane >> 4, ln = lane & 15;
  const int qt = blockIdx.x;
  const int b  = blockIdx.y / NHH, h = blockIdx.y % NHH;
  const int q0 = qt * 64;
  const size_t rowbase = (size_t)b * LLN;
  const int hc = h * DH;
  const float scale = 0.125f;  // 1/sqrt(64)

  for (int i = tid; i < 64 * 64; i += 128) {
    int r = i >> 6, c = i & 63;
    Qs[r * 66 + c] = f2bf(Q[(rowbase + q0 + r) * DDIM + hc + c]);
  }

  float rowM[8], rowL[8];
  v8f oacc[4];
#pragma unroll
  for (int r = 0; r < 8; ++r) { rowM[r] = -3.0e38f; rowL[r] = 0.0f; }
#pragma unroll
  for (int nt = 0; nt < 4; ++nt)
#pragma unroll
    for (int r = 0; r < 8; ++r) oacc[nt][r] = 0.0f;

  for (int t = 0; t <= qt; ++t) {
    int k0 = t * 64;
    __syncthreads();
    for (int i = tid; i < 64 * 64; i += 128) {
      int r = i >> 6, c = i & 63;
      Ks[r * 66 + c] = f2bf(K[(rowbase + k0 + r) * DDIM + hc + c]);
      Vs[c * 66 + r] = f2bf(V[(rowbase + k0 + r) * DDIM + hc + c]);
    }
    __syncthreads();

    // ---- S = (Q Kt) for this wave's 16x64 strip ----
    v8f s[4];
#pragma unroll
    for (int nt = 0; nt < 4; ++nt)
#pragma unroll
      for (int r = 0; r < 8; ++r) s[nt][r] = 0.0f;
#pragma unroll
    for (int kk = 0; kk < 64; kk += 32) {
      v16bf af; uint32_t* au = reinterpret_cast<uint32_t*>(&af);
      int m = 16 * wave + ln;
#pragma unroll
      for (int v = 0; v < 8; ++v) {
        int kq = kk + 2 * v + 16 * g;
        au[v] = *reinterpret_cast<const uint32_t*>(&Qs[m * 66 + kq]);
      }
#pragma unroll
      for (int nt = 0; nt < 4; ++nt) {
        v16bf bf; uint32_t* bu = reinterpret_cast<uint32_t*>(&bf);
        int n = 16 * nt + ln;
#pragma unroll
        for (int v = 0; v < 8; ++v) {
          int kq = kk + 2 * v + 16 * g;
          bu[v] = *reinterpret_cast<const uint32_t*>(&Ks[n * 66 + kq]);
        }
        s[nt] = __builtin_amdgcn_wmma_f32_16x16x32_bf16(
            false, af, false, bf, (short)0, s[nt], false, false);
      }
    }

    // ---- scale, causal mask, online softmax ----
#pragma unroll
    for (int r = 0; r < 8; ++r) {
      int qrow = q0 + 16 * wave + r + 8 * g;
      float mx = -3.0e38f;
#pragma unroll
      for (int nt = 0; nt < 4; ++nt) {
        int key = k0 + 16 * nt + ln;
        float v = s[nt][r] * scale;
        if (key > qrow) v = -1.0e9f;  // matches reference mask*-1e9
        s[nt][r] = v;
        mx = fmaxf(mx, v);
      }
#pragma unroll
      for (int off = 1; off < 16; off <<= 1)
        mx = fmaxf(mx, __shfl_xor(mx, off, 32));  // within 16-lane half
      float newM = fmaxf(rowM[r], mx);
      float fac  = __expf(rowM[r] - newM);
      rowM[r] = newM;
      float lsum = 0.0f;
#pragma unroll
      for (int nt = 0; nt < 4; ++nt) {
        float p = __expf(s[nt][r] - newM);
        s[nt][r] = p;
        lsum += p;
      }
#pragma unroll
      for (int off = 1; off < 16; off <<= 1)
        lsum += __shfl_xor(lsum, off, 32);
      rowL[r] = rowL[r] * fac + lsum;
#pragma unroll
      for (int nt = 0; nt < 4; ++nt) oacc[nt][r] *= fac;
      int m = 16 * wave + r + 8 * g;
#pragma unroll
      for (int nt = 0; nt < 4; ++nt)
        Ps[m * 66 + 16 * nt + ln] = f2bf(s[nt][r]);
    }

    // ---- O += P @ V (DS ops in-order per wave; strip is wave-private) ----
#pragma unroll
    for (int kk = 0; kk < 64; kk += 32) {
      v16bf af; uint32_t* au = reinterpret_cast<uint32_t*>(&af);
      int m = 16 * wave + ln;
#pragma unroll
      for (int v = 0; v < 8; ++v) {
        int kq = kk + 2 * v + 16 * g;
        au[v] = *reinterpret_cast<const uint32_t*>(&Ps[m * 66 + kq]);
      }
#pragma unroll
      for (int nt = 0; nt < 4; ++nt) {
        v16bf bf; uint32_t* bu = reinterpret_cast<uint32_t*>(&bf);
        int n = 16 * nt + ln;  // dh dim
#pragma unroll
        for (int v = 0; v < 8; ++v) {
          int kq = kk + 2 * v + 16 * g;
          bu[v] = *reinterpret_cast<const uint32_t*>(&Vs[n * 66 + kq]);
        }
        oacc[nt] = __builtin_amdgcn_wmma_f32_16x16x32_bf16(
            false, af, false, bf, (short)0, oacc[nt], false, false);
      }
    }
  }

#pragma unroll
  for (int nt = 0; nt < 4; ++nt)
#pragma unroll
    for (int r = 0; r < 8; ++r) {
      int qrow = q0 + 16 * wave + r + 8 * g;
      O[(rowbase + qrow) * DDIM + hc + 16 * nt + ln] = oacc[nt][r] / rowL[r];
    }
}

// =============================================================================
// Ring attention: per token, 5 keys {h[l+1], h[l], h[l-1], e[l], s}.
// One wave per token; lane covers 2 dims of each 64-dim head. Memory-bound.
// =============================================================================
__global__ __launch_bounds__(256) void ring_attn(
    const float* __restrict__ Qh, const float* __restrict__ Kh,
    const float* __restrict__ Vh, const float* __restrict__ Ke,
    const float* __restrict__ Ve, const float* __restrict__ ks,
    const float* __restrict__ vs, float* __restrict__ O)
{
  int wave = threadIdx.x >> 5, lane = threadIdx.x & 31;
  int pos = blockIdx.x * 8 + wave;  // 0..BLR-1
  int b = pos / LLN, l = pos % LLN;
  int lp = (l + 1) % LLN, lm = (l - 1 + LLN) % LLN;
  const float* kptr[5];
  const float* vptr[5];
  kptr[0] = Kh + ((size_t)b * LLN + lp) * DDIM;
  kptr[1] = Kh + ((size_t)b * LLN + l)  * DDIM;
  kptr[2] = Kh + ((size_t)b * LLN + lm) * DDIM;
  kptr[3] = Ke + ((size_t)b * LLN + l)  * DDIM;
  kptr[4] = ks + (size_t)b * DDIM;
  vptr[0] = Vh + ((size_t)b * LLN + lp) * DDIM;
  vptr[1] = Vh + ((size_t)b * LLN + l)  * DDIM;
  vptr[2] = Vh + ((size_t)b * LLN + lm) * DDIM;
  vptr[3] = Ve + ((size_t)b * LLN + l)  * DDIM;
  vptr[4] = vs + (size_t)b * DDIM;
  const float* q = Qh + (size_t)pos * DDIM;
  float* o = O + (size_t)pos * DDIM;

  for (int hd = 0; hd < NHH; ++hd) {
    int d0 = hd * DH + lane * 2;
    float q0 = q[d0], q1 = q[d0 + 1];
    float lg[5];
#pragma unroll
    for (int j = 0; j < 5; ++j) {
      float p = q0 * kptr[j][d0] + q1 * kptr[j][d0 + 1];
#pragma unroll
      for (int off = 1; off < 32; off <<= 1) p += __shfl_xor(p, off, 32);
      lg[j] = p * 0.125f;
    }
    float mx = lg[0];
#pragma unroll
    for (int j = 1; j < 5; ++j) mx = fmaxf(mx, lg[j]);
    float sum = 0.0f, w[5];
#pragma unroll
    for (int j = 0; j < 5; ++j) { w[j] = __expf(lg[j] - mx); sum += w[j]; }
    float inv = 1.0f / sum;
    float o0 = 0.0f, o1 = 0.0f;
#pragma unroll
    for (int j = 0; j < 5; ++j) {
      o0 += w[j] * vptr[j][d0];
      o1 += w[j] * vptr[j][d0 + 1];
    }
    o[d0] = o0 * inv;
    o[d0 + 1] = o1 * inv;
  }
}

// =============================================================================
// s-update attention: query s (1 token/batch), keys m_c = [s, h_new, h2]
// (2049 keys). One block per (b, head). Logits cached in LDS.
// =============================================================================
__global__ __launch_bounds__(256) void s_attn(
    const float* __restrict__ qs, const float* __restrict__ ksx,
    const float* __restrict__ Kh, const float* __restrict__ Kh2,
    const float* __restrict__ vsx, const float* __restrict__ Vh,
    const float* __restrict__ Vh2, float* __restrict__ O)
{
  constexpr int NKEY = 2 * LLN + 1;
  __shared__ float qsh[DH];
  __shared__ float wgt[NKEY];
  __shared__ float red[8];
  __shared__ float osum[4][DH];
  __shared__ float smax_sh, ssum_sh;
  int b = blockIdx.x / NHH, h = blockIdx.x % NHH;
  int tid = threadIdx.x, lane = tid & 31, wave = tid >> 5;
  if (tid < DH) qsh[tid] = qs[(size_t)b * DDIM + h * DH + tid];
  __syncthreads();

  float lmax = -3.0e38f;
  for (int key = tid; key < NKEY; key += 256) {
    const float* kp;
    if (key == 0)        kp = ksx + (size_t)b * DDIM + h * DH;
    else if (key <= LLN) kp = Kh  + ((size_t)b * LLN + (key - 1)) * DDIM + h * DH;
    else                 kp = Kh2 + ((size_t)b * LLN + (key - 1 - LLN)) * DDIM + h * DH;
    float acc = 0.0f;
#pragma unroll 16
    for (int d = 0; d < DH; ++d) acc += qsh[d] * kp[d];
    acc *= 0.125f;
    wgt[key] = acc;
    lmax = fmaxf(lmax, acc);
  }
  for (int off = 1; off < 32; off <<= 1) lmax = fmaxf(lmax, __shfl_xor(lmax, off, 32));
  if (lane == 0) red[wave] = lmax;
  __syncthreads();
  if (tid == 0) {
    float m = red[0];
    for (int i = 1; i < 8; ++i) m = fmaxf(m, red[i]);
    smax_sh = m;
  }
  __syncthreads();
  float smax = smax_sh;
  float lsum = 0.0f;
  for (int key = tid; key < NKEY; key += 256) {
    float w = __expf(wgt[key] - smax);
    wgt[key] = w;
    lsum += w;
  }
  for (int off = 1; off < 32; off <<= 1) lsum += __shfl_xor(lsum, off, 32);
  if (lane == 0) red[wave] = lsum;
  __syncthreads();
  if (tid == 0) {
    float m = 0.0f;
    for (int i = 0; i < 8; ++i) m += red[i];
    ssum_sh = m;
  }
  __syncthreads();
  float inv = 1.0f / ssum_sh;
  int d = tid & 63, grp = tid >> 6;
  float acc = 0.0f;
  for (int key = grp; key < NKEY; key += 4) {
    const float* vp;
    if (key == 0)        vp = vsx + (size_t)b * DDIM + h * DH;
    else if (key <= LLN) vp = Vh  + ((size_t)b * LLN + (key - 1)) * DDIM + h * DH;
    else                 vp = Vh2 + ((size_t)b * LLN + (key - 1 - LLN)) * DDIM + h * DH;
    acc += wgt[key] * vp[d];
  }
  osum[grp][d] = acc;
  __syncthreads();
  if (tid < DH)
    O[(size_t)b * DDIM + h * DH + tid] =
        (osum[0][tid] + osum[1][tid] + osum[2][tid] + osum[3][tid]) * inv;
}

// =============================================================================
// LayerNorm(X + R) * gamma + beta, one wave per 512-elem row.
// =============================================================================
__global__ __launch_bounds__(256) void layernorm_res(
    const float* __restrict__ X, const float* __restrict__ R,
    const float* __restrict__ gamma, const float* __restrict__ beta,
    float* __restrict__ Out, int Mrows)
{
  int wave = threadIdx.x >> 5, lane = threadIdx.x & 31;
  int row = blockIdx.x * 8 + wave;
  if (row >= Mrows) return;
  const float* x = X + (size_t)row * DDIM;
  const float* r = R ? (R + (size_t)row * DDIM) : nullptr;
  float vals[16];
  float sum = 0.0f;
#pragma unroll
  for (int i = 0; i < 16; ++i) {
    int c = lane + 32 * i;
    float v = x[c] + (r ? r[c] : 0.0f);
    vals[i] = v;
    sum += v;
  }
  for (int off = 1; off < 32; off <<= 1) sum += __shfl_xor(sum, off, 32);
  float mean = sum * (1.0f / DDIM);
  float vsum = 0.0f;
#pragma unroll
  for (int i = 0; i < 16; ++i) {
    float dv = vals[i] - mean;
    vsum += dv * dv;
  }
  for (int off = 1; off < 32; off <<= 1) vsum += __shfl_xor(vsum, off, 32);
  float invstd = rsqrtf(vsum * (1.0f / DDIM) + 1e-6f);
#pragma unroll
  for (int i = 0; i < 16; ++i) {
    int c = lane + 32 * i;
    Out[(size_t)row * DDIM + c] = (vals[i] - mean) * invstd * gamma[c] + beta[c];
  }
}

__global__ void mean_rows(const float* __restrict__ E, float* __restrict__ S) {
  int b = blockIdx.y;
  int d = blockIdx.x * 256 + threadIdx.x;
  float acc = 0.0f;
  for (int l = 0; l < LLN; ++l) acc += E[((size_t)b * LLN + l) * DDIM + d];
  S[(size_t)b * DDIM + d] = acc * (1.0f / LLN);
}

__global__ void copy_f32(const float* __restrict__ src, float* __restrict__ dst, int n) {
  int i = blockIdx.x * 256 + threadIdx.x;
  if (i < n) dst[i] = src[i];
}

// =============================================================================
// Host orchestration
// =============================================================================
extern "C" void kernel_launch(void* const* d_in, const int* in_sizes, int n_in,
                              void* d_out, int out_size, void* d_ws, size_t ws_size,
                              hipStream_t stream) {
  (void)in_sizes; (void)n_in; (void)out_size; (void)ws_size;
  const float* tar  = (const float*)d_in[0];
  const float* e    = (const float*)d_in[1];
  // d_in[2] look_ahead_mask: causal structure applied analytically
  const float* wq_t = (const float*)d_in[3];
  const float* wk_t = (const float*)d_in[4];
  const float* wv_t = (const float*)d_in[5];
  const float* wo_t = (const float*)d_in[6];
  const float* bo_t = (const float*)d_in[7];
  const float* wq_s = (const float*)d_in[8];
  const float* wk_s = (const float*)d_in[9];
  const float* wv_s = (const float*)d_in[10];
  const float* wo_s = (const float*)d_in[11];
  const float* bo_s = (const float*)d_in[12];
  const float* w1   = (const float*)d_in[13];
  const float* b1   = (const float*)d_in[14];
  const float* w2   = (const float*)d_in[15];
  const float* b2   = (const float*)d_in[16];
  const float* g1   = (const float*)d_in[17];
  const float* be1  = (const float*)d_in[18];
  const float* g2   = (const float*)d_in[19];
  const float* be2  = (const float*)d_in[20];
  // d_in[21] cycle_num==2, d_in[22] num_heads==8 (fixed by setup; device
  // scalars can't steer launch counts under graph capture, so hardcoded).

  float* out2 = (float*)d_out;                       // (B,L,D)
  float* sout = (float*)d_out + (size_t)BLR * DDIM;  // (B,D)

  char* ws = (char*)d_ws;
  const size_t SL = (size_t)BLR * DDIM * sizeof(float);  // 16 MB slab
  float* h2   = (float*)(ws + 0 * SL);
  float* Ke   = (float*)(ws + 1 * SL);
  float* Ve   = (float*)(ws + 2 * SL);
  float* Kh2  = (float*)(ws + 3 * SL);
  float* Vh2  = (float*)(ws + 4 * SL);
  float* bufQ = (float*)(ws + 5 * SL);
  float* bufK = (float*)(ws + 6 * SL);
  float* bufV = (float*)(ws + 7 * SL);
  float* bufO = (float*)(ws + 8 * SL);
  float* bufH = (float*)(ws + 9 * SL);
  float* out1 = (float*)(ws + 10 * SL);
  float* t1   = (float*)(ws + 1 * SL);  // 64 MB FFN tmp, overlaps Ke..Vh2 (dead by then)
  float* sA   = (float*)(ws + 11 * SL);
  float* qs   = sA  + BB * DDIM;
  float* ksb  = qs  + BB * DDIM;
  float* vsb  = ksb + BB * DDIM;
  float* sO   = vsb + BB * DDIM;

  const dim3 blk(256);
  const dim3 g_proj(DDIM / 128, BLR / 128);   // (4, 64)
  const dim3 g_ffn1(DFFN / 128, BLR / 128);   // (16, 64)
  const dim3 g_s(DDIM / 128, 1);              // M=8 rows

  // ---------------- Phase A: causal self-attention on tar ----------------
  gemm_wmma<false><<<g_proj, blk, 0, stream>>>(tar, wq_t, nullptr, bufQ, BLR, DDIM, DDIM);
  gemm_wmma<false><<<g_proj, blk, 0, stream>>>(tar, wk_t, nullptr, bufK, BLR, DDIM, DDIM);
  gemm_wmma<false><<<g_proj, blk, 0, stream>>>(tar, wv_t, nullptr, bufV, BLR, DDIM, DDIM);
  flash_attn_causal<<<dim3(LLN / 64, BB * NHH), dim3(128), 0, stream>>>(bufQ, bufK, bufV, bufO);
  gemm_wmma<false><<<g_proj, blk, 0, stream>>>(bufO, wo_t, bo_t, bufH, BLR, DDIM, DDIM);
  layernorm_res<<<BLR / 8, blk, 0, stream>>>(bufH, tar, g1, be1, h2, BLR);

  // ---------------- Star-loop invariant K/V projections ----------------
  gemm_wmma<false><<<g_proj, blk, 0, stream>>>(e,  wk_s, nullptr, Ke,  BLR, DDIM, DDIM);
  gemm_wmma<false><<<g_proj, blk, 0, stream>>>(e,  wv_s, nullptr, Ve,  BLR, DDIM, DDIM);
  gemm_wmma<false><<<g_proj, blk, 0, stream>>>(h2, wk_s, nullptr, Kh2, BLR, DDIM, DDIM);
  gemm_wmma<false><<<g_proj, blk, 0, stream>>>(h2, wv_s, nullptr, Vh2, BLR, DDIM, DDIM);
  mean_rows<<<dim3(DDIM / 256, BB), blk, 0, stream>>>(e, sA);

  const float* curH  = e;    // h = e initially
  const float* curKh = Ke;   // h==e -> Kh==Ke, Vh==Ve for cycle 0
  const float* curVh = Ve;

  for (int cyc = 0; cyc < 2; ++cyc) {
    // projections of current s (used by both attentions this cycle)
    gemm_wmma<false><<<g_s, blk, 0, stream>>>(sA, wq_s, nullptr, qs,  BB, DDIM, DDIM);
    gemm_wmma<false><<<g_s, blk, 0, stream>>>(sA, wk_s, nullptr, ksb, BB, DDIM, DDIM);
    gemm_wmma<false><<<g_s, blk, 0, stream>>>(sA, wv_s, nullptr, vsb, BB, DDIM, DDIM);
    // query projection of current h
    gemm_wmma<false><<<g_proj, blk, 0, stream>>>(curH, wq_s, nullptr, bufQ, BLR, DDIM, DDIM);
    // 5-key ring attention; then output projection + relu -> new h
    ring_attn<<<BLR / 8, blk, 0, stream>>>(bufQ, curKh, curVh, Ke, Ve, ksb, vsb, bufO);
    gemm_wmma<true><<<g_proj, blk, 0, stream>>>(bufO, wo_s, bo_s, bufH, BLR, DDIM, DDIM);
    // K/V of new h (used by s-attention now, ring attention next cycle)
    gemm_wmma<false><<<g_proj, blk, 0, stream>>>(bufH, wk_s, nullptr, bufK, BLR, DDIM, DDIM);
    gemm_wmma<false><<<g_proj, blk, 0, stream>>>(bufH, wv_s, nullptr, bufV, BLR, DDIM, DDIM);
    // s update over m_c = [s_old, h_new, h2]
    s_attn<<<BB * NHH, blk, 0, stream>>>(qs, ksb, bufK, Kh2, vsb, bufV, Vh2, sO);
    gemm_wmma<true><<<g_s, blk, 0, stream>>>(sO, wo_s, bo_s, sA, BB, DDIM, DDIM);
    curH = bufH; curKh = bufK; curVh = bufV;
  }

  // ---------------- Output head: LN + FFN + LN ----------------
  layernorm_res<<<BLR / 8, blk, 0, stream>>>(curH, e, g1, be1, out1, BLR);
  gemm_wmma<true ><<<g_ffn1, blk, 0, stream>>>(out1, w1, b1, t1, BLR, DFFN, DDIM);
  gemm_wmma<false><<<g_proj, blk, 0, stream>>>(t1, w2, b2, bufO, BLR, DDIM, DFFN);
  layernorm_res<<<BLR / 8, blk, 0, stream>>>(bufO, out1, g2, be2, out2, BLR);
  copy_f32<<<(BB * DDIM + 255) / 256, blk, 0, stream>>>(sA, sout, BB * DDIM);
}
